// EchoStateTransformer_67972152427211
// MI455X (gfx1250) — compile-verified, matrix operand order checked by reading, not run on hardware
//
#include <hip/hip_runtime.h>
#include <hip/hip_bf16.h>
#include <math.h>

// ---------------------------------------------------------------------------
// EchoStateTransformer forward for MI455X (gfx1250, wave32, WMMA).
// B=1024, I=512, O=512, M=8, R=512, A=16, D=128.
// Heavy GEMMs on v_wmma_f32_16x16x32_bf16 (f32 accumulate), 2 N-tiles/wave.
// ---------------------------------------------------------------------------

#define BB   1024
#define II   512
#define OO   512
#define MM   8
#define RR   512
#define AA   16
#define DD   128   // M*A

typedef __attribute__((ext_vector_type(16))) __bf16 v16bf;
typedef __attribute__((ext_vector_type(8)))  float  v8f;

__device__ __forceinline__ __bf16 cvt_bf16(float f) {
    return (__bf16)f;   // native bf16 convert on gfx1250
}

// A fragment: 16x32 bf16 per wave (ISA 16-bit A layout).
// lane 0-15: row M=lane, K = {kb..kb+7, kb+16..kb+23} with kb=0
// lane 16-31: row M=lane-16, same pattern with kb=8
// ap points at &A[row*lda + k0 + kb]; 32B aligned -> float4 loads.
__device__ __forceinline__ v16bf load_a_frag(const float* ap) {
    const float4* p = reinterpret_cast<const float4*>(ap);
    float4 x0 = p[0], x1 = p[1], x2 = p[4], x3 = p[5];
    v16bf a;
    a[0]  = cvt_bf16(x0.x); a[1]  = cvt_bf16(x0.y);
    a[2]  = cvt_bf16(x0.z); a[3]  = cvt_bf16(x0.w);
    a[4]  = cvt_bf16(x1.x); a[5]  = cvt_bf16(x1.y);
    a[6]  = cvt_bf16(x1.z); a[7]  = cvt_bf16(x1.w);
    a[8]  = cvt_bf16(x2.x); a[9]  = cvt_bf16(x2.y);
    a[10] = cvt_bf16(x2.z); a[11] = cvt_bf16(x2.w);
    a[12] = cvt_bf16(x3.x); a[13] = cvt_bf16(x3.y);
    a[14] = cvt_bf16(x3.z); a[15] = cvt_bf16(x3.w);
    return a;
}

// B fragment from LDS tile stored [n][k] (n=0..15, k=0..31, bf16).
__device__ __forceinline__ v16bf load_b_frag(const __bf16* Bs, int lane) {
    const __bf16* bp = Bs + (lane & 15) * 32 + (((lane >> 4) & 1) << 3);
    v16bf b;
#pragma unroll
    for (int i = 0; i < 8; ++i) {
        b[i]     = bp[i];
        b[i + 8] = bp[i + 16];
    }
    return b;
}

#define WMMA_BF16(a, b, c) \
    __builtin_amdgcn_wmma_f32_16x16x32_bf16(false, (a), false, (b), (short)0, (c), false, false)

// ---------------------------------------------------------------------------
// Kernel 0: spectral radius per unit via power iteration.
// scale[m] = sr[m] / max|eig(W_m)|  (estimate)
// ---------------------------------------------------------------------------
__global__ __launch_bounds__(256) void esn_spectral_kernel(
    const float* __restrict__ W, const float* __restrict__ sr,
    float* __restrict__ scale) {
    __shared__ float v[RR], y[RR], red[256];
    const int m = blockIdx.x;
    const int t = threadIdx.x;
    const float* Wm = W + (size_t)m * RR * RR;

    v[t] = 1.0f; v[t + 256] = 1.0f;
    __syncthreads();

    float lam_acc = 0.0f;
    for (int it = 0; it < 32; ++it) {
#pragma unroll
        for (int rr = 0; rr < 2; ++rr) {
            int r = t + rr * 256;
            const float4* row = reinterpret_cast<const float4*>(Wm + (size_t)r * RR);
            float s = 0.0f;
            for (int c = 0; c < RR / 4; ++c) {
                float4 w = row[c];
                s += w.x * v[4 * c] + w.y * v[4 * c + 1] +
                     w.z * v[4 * c + 2] + w.w * v[4 * c + 3];
            }
            y[r] = s;
        }
        __syncthreads();
        red[t] = y[t] * y[t] + y[t + 256] * y[t + 256];
        __syncthreads();
        for (int o = 128; o > 0; o >>= 1) {
            if (t < o) red[t] += red[t + o];
            __syncthreads();
        }
        float lam = sqrtf(red[0]);
        if (it >= 24) lam_acc += lam;
        float inv = (lam > 1e-30f) ? 1.0f / lam : 0.0f;
        v[t] = y[t] * inv; v[t + 256] = y[t + 256] * inv;
        __syncthreads();
    }
    if (t == 0) {
        float lam = fmaxf(lam_acc * (1.0f / 8.0f), 1e-12f);
        scale[m] = sr[m] / lam;
    }
}

// ---------------------------------------------------------------------------
// Kernel 1: Q[b, m*16+a] = sum_i Xi[b,i] * Wq[m,i,a]
// grid (B/64, D/32), block 128; each wave: 16 rows x 32 cols (2 WMMA tiles)
// ---------------------------------------------------------------------------
__global__ __launch_bounds__(128) void esn_q_kernel(
    const float* __restrict__ Xi, const float* __restrict__ Wq,
    float* __restrict__ Qbuf) {
    __shared__ __bf16 Bs[2][512];
    const int t = threadIdx.x, lane = t & 31, wave = t >> 5;
    const int row0 = blockIdx.x * 64 + wave * 16;
    const int n0 = blockIdx.y * 32;

    const int am = row0 + (lane & 15);
    const int kb = ((lane >> 4) & 1) << 3;
    v8f acc0 = {0.f, 0.f, 0.f, 0.f, 0.f, 0.f, 0.f, 0.f};
    v8f acc1 = acc0;

    for (int k0 = 0; k0 < II; k0 += 32) {
#pragma unroll
        for (int u = 0; u < 8; ++u) {
            int e = t + u * 128;
            int n2 = e & 31, k = e >> 5;
            int n = n0 + n2;   // column in [0,128): unit = n>>4, a = n&15
            float g = Wq[((size_t)(n >> 4) * II + (k0 + k)) * AA + (n & 15)];
            Bs[n2 >> 4][(n2 & 15) * 32 + k] = cvt_bf16(g);
        }
        __syncthreads();
        v16bf a  = load_a_frag(Xi + (size_t)am * II + k0 + kb);
        v16bf b0 = load_b_frag(Bs[0], lane);
        v16bf b1 = load_b_frag(Bs[1], lane);
        acc0 = WMMA_BF16(a, b0, acc0);
        acc1 = WMMA_BF16(a, b1, acc1);
        __syncthreads();
    }
    const int rb = row0 + (((lane >> 4) & 1) << 3);
    const int cn = n0 + (lane & 15);
#pragma unroll
    for (int j = 0; j < 8; ++j) {
        Qbuf[(size_t)(rb + j) * DD + cn]      = acc0[j];
        Qbuf[(size_t)(rb + j) * DD + cn + 16] = acc1[j];
    }
}

// ---------------------------------------------------------------------------
// Kernel 2: K/V:  Kbuf[n][b][m*16+a] = sum_r Si[b,n,r]*Wk[m,r,a]  (V alike)
// grid (B/64, D/16, M(n)), block 128. One A-frag feeds two WMMAs (K and V).
// ---------------------------------------------------------------------------
__global__ __launch_bounds__(128) void esn_kv_kernel(
    const float* __restrict__ Si, const float* __restrict__ Wk,
    const float* __restrict__ Wv, float* __restrict__ Kbuf,
    float* __restrict__ Vbuf) {
    __shared__ __bf16 Bk[512], Bv[512];
    const int t = threadIdx.x, lane = t & 31, wave = t >> 5;
    const int row0 = blockIdx.x * 64 + wave * 16;
    const int n0 = blockIdx.y * 16;
    const int nu = blockIdx.z;                        // Si unit index ('n')
    const size_t uoff = (size_t)(n0 >> 4) * RR * AA;  // Wk/Wv unit m = n0/16
    const float* BKb = Wk + uoff;
    const float* BVb = Wv + uoff;

    const int am = row0 + (lane & 15);
    const int kb = ((lane >> 4) & 1) << 3;
    v8f accK = {0.f, 0.f, 0.f, 0.f, 0.f, 0.f, 0.f, 0.f};
    v8f accV = accK;

    for (int k0 = 0; k0 < RR; k0 += 32) {
#pragma unroll
        for (int u = 0; u < 4; ++u) {
            int e = t + u * 128;
            int n = e & 15, k = e >> 4;
            size_t gi = (size_t)(k0 + k) * AA + n;
            Bk[n * 32 + k] = cvt_bf16(BKb[gi]);
            Bv[n * 32 + k] = cvt_bf16(BVb[gi]);
        }
        __syncthreads();
        v16bf a  = load_a_frag(Si + (size_t)am * (MM * RR) + (size_t)nu * RR + k0 + kb);
        v16bf bk = load_b_frag(Bk, lane);
        v16bf bv = load_b_frag(Bv, lane);
        accK = WMMA_BF16(a, bk, accK);
        accV = WMMA_BF16(a, bv, accV);
        __syncthreads();
    }
    const int cn = n0 + (lane & 15);
    const int rb = row0 + (((lane >> 4) & 1) << 3);
    const size_t base = (size_t)nu * BB * DD;
#pragma unroll
    for (int j = 0; j < 8; ++j) {
        size_t idx = base + (size_t)(rb + j) * DD + cn;
        Kbuf[idx] = accK[j];
        Vbuf[idx] = accV[j];
    }
}

// ---------------------------------------------------------------------------
// Kernel 3: attention (scores -> softmax/sqrt(D) -> Ui). Tiny dims: VALU.
// One thread per (b, m, q).  Ui[b][m][q*16+a]
// ---------------------------------------------------------------------------
__global__ __launch_bounds__(256) void esn_attn_kernel(
    const float* __restrict__ Qb, const float* __restrict__ Kb,
    const float* __restrict__ Vb, float* __restrict__ Ui) {
    int id = blockIdx.x * blockDim.x + threadIdx.x;
    if (id >= BB * MM * MM) return;
    const int b = id >> 6;
    const int m = (id >> 3) & 7;
    const int q = id & 7;

    float qv[AA];
    const float* qp = Qb + (size_t)b * DD + q * AA;
#pragma unroll
    for (int a = 0; a < AA; ++a) qv[a] = qp[a];

    float s[MM];
    float mx = -1e30f;
#pragma unroll
    for (int n = 0; n < MM; ++n) {
        const float* kp = Kb + (size_t)n * BB * DD + (size_t)b * DD + m * AA;
        float acc = 0.0f;
#pragma unroll
        for (int a = 0; a < AA; ++a) acc += qv[a] * kp[a];
        s[n] = acc;
        mx = fmaxf(mx, acc);
    }
    float den = 0.0f;
#pragma unroll
    for (int n = 0; n < MM; ++n) { s[n] = __expf(s[n] - mx); den += s[n]; }
    const float invd = 1.0f / (den * sqrtf((float)DD));  // softmax then /sqrt(D)

    float o[AA];
#pragma unroll
    for (int a = 0; a < AA; ++a) o[a] = 0.0f;
#pragma unroll
    for (int n = 0; n < MM; ++n) {
        const float w = s[n] * invd;
        const float* vp = Vb + (size_t)n * BB * DD + (size_t)b * DD + m * AA;
#pragma unroll
        for (int a = 0; a < AA; ++a) o[a] += w * vp[a];
    }
    float* up = Ui + (size_t)b * (MM * DD) + m * DD + q * AA;
#pragma unroll
    for (int a = 0; a < AA; ++a) up[a] = o[a];
}

// ---------------------------------------------------------------------------
// Kernel 4: state update per unit m. Fused K=640 GEMM, 2 N-tiles per wave:
//   pre = Ui[b,m,:]*Win[m] + Si[b,m,:]*(W[m]*scale[m]) + bias[m]
//   Snew = (1-lr[m])*Si + lr[m]*tanh(pre)
// grid (B/64, R/32, M), block 128
// ---------------------------------------------------------------------------
__global__ __launch_bounds__(128) void esn_update_kernel(
    const float* __restrict__ Si, const float* __restrict__ Ui,
    const float* __restrict__ Win, const float* __restrict__ W,
    const float* __restrict__ scale, const float* __restrict__ bias,
    const float* __restrict__ lr, float* __restrict__ Snew) {
    __shared__ __bf16 Bs[2][512];
    const int t = threadIdx.x, lane = t & 31, wave = t >> 5;
    const int row0 = blockIdx.x * 64 + wave * 16;
    const int n0 = blockIdx.y * 32;
    const int m = blockIdx.z;
    const float sc = scale[m];

    const int am = row0 + (lane & 15);
    const int kb = ((lane >> 4) & 1) << 3;
    v8f acc0 = {0.f, 0.f, 0.f, 0.f, 0.f, 0.f, 0.f, 0.f};
    v8f acc1 = acc0;

    // segment 1: feed = Ui[b,m,0:128] @ Win[m] (128 x 512)
    const float* B1 = Win + (size_t)m * DD * RR;
    for (int k0 = 0; k0 < DD; k0 += 32) {
#pragma unroll
        for (int u = 0; u < 8; ++u) {
            int e = t + u * 128;
            int n2 = e & 31, k = e >> 5;
            Bs[n2 >> 4][(n2 & 15) * 32 + k] =
                cvt_bf16(B1[(size_t)(k0 + k) * RR + n0 + n2]);
        }
        __syncthreads();
        v16bf a  = load_a_frag(Ui + (size_t)am * (MM * DD) + m * DD + k0 + kb);
        v16bf b0 = load_b_frag(Bs[0], lane);
        v16bf b1 = load_b_frag(Bs[1], lane);
        acc0 = WMMA_BF16(a, b0, acc0);
        acc1 = WMMA_BF16(a, b1, acc1);
        __syncthreads();
    }
    // segment 2: echo = Si[b,m,0:512] @ (W[m]*sc) (512 x 512)
    const float* B2 = W + (size_t)m * RR * RR;
    for (int k0 = 0; k0 < RR; k0 += 32) {
#pragma unroll
        for (int u = 0; u < 8; ++u) {
            int e = t + u * 128;
            int n2 = e & 31, k = e >> 5;
            Bs[n2 >> 4][(n2 & 15) * 32 + k] =
                cvt_bf16(sc * B2[(size_t)(k0 + k) * RR + n0 + n2]);
        }
        __syncthreads();
        v16bf a  = load_a_frag(Si + (size_t)am * (MM * RR) + m * RR + k0 + kb);
        v16bf b0 = load_b_frag(Bs[0], lane);
        v16bf b1 = load_b_frag(Bs[1], lane);
        acc0 = WMMA_BF16(a, b0, acc0);
        acc1 = WMMA_BF16(a, b1, acc1);
        __syncthreads();
    }
    // epilogue
    const float lrm = lr[m];
    const int rb = row0 + (((lane >> 4) & 1) << 3);
#pragma unroll
    for (int tile = 0; tile < 2; ++tile) {
        const int r = n0 + tile * 16 + (lane & 15);
        const float bm = bias[(size_t)m * RR + r];
        const v8f& acc = tile ? acc1 : acc0;
#pragma unroll
        for (int j = 0; j < 8; ++j) {
            size_t sidx = (size_t)(rb + j) * (MM * RR) + (size_t)m * RR + r;
            float sold = Si[sidx];
            float x = acc[j] + bm;
            Snew[sidx] = (1.0f - lrm) * sold + lrm * tanhf(x);
        }
    }
}

// ---------------------------------------------------------------------------
// Kernel 5: Yi = Snew[B, M*R] @ Wout[M*R, O], 2 N-tiles per wave
// grid (B/64, O/32), block 128
// ---------------------------------------------------------------------------
__global__ __launch_bounds__(128) void esn_out_kernel(
    const float* __restrict__ Snew, const float* __restrict__ Wout,
    float* __restrict__ Yi) {
    __shared__ __bf16 Bs[2][512];
    const int t = threadIdx.x, lane = t & 31, wave = t >> 5;
    const int row0 = blockIdx.x * 64 + wave * 16;
    const int n0 = blockIdx.y * 32;

    const int am = row0 + (lane & 15);
    const int kb = ((lane >> 4) & 1) << 3;
    v8f acc0 = {0.f, 0.f, 0.f, 0.f, 0.f, 0.f, 0.f, 0.f};
    v8f acc1 = acc0;

    for (int k0 = 0; k0 < MM * RR; k0 += 32) {
#pragma unroll
        for (int u = 0; u < 8; ++u) {
            int e = t + u * 128;
            int n2 = e & 31, k = e >> 5;
            Bs[n2 >> 4][(n2 & 15) * 32 + k] =
                cvt_bf16(Wout[(size_t)(k0 + k) * OO + n0 + n2]);
        }
        __syncthreads();
        v16bf a  = load_a_frag(Snew + (size_t)am * (MM * RR) + k0 + kb);
        v16bf b0 = load_b_frag(Bs[0], lane);
        v16bf b1 = load_b_frag(Bs[1], lane);
        acc0 = WMMA_BF16(a, b0, acc0);
        acc1 = WMMA_BF16(a, b1, acc1);
        __syncthreads();
    }
    const int rb = row0 + (((lane >> 4) & 1) << 3);
    const int cn = n0 + (lane & 15);
#pragma unroll
    for (int j = 0; j < 8; ++j) {
        Yi[(size_t)(rb + j) * OO + cn]      = acc0[j];
        Yi[(size_t)(rb + j) * OO + cn + 16] = acc1[j];
    }
}

// ---------------------------------------------------------------------------
// Host-side launcher
// ---------------------------------------------------------------------------
extern "C" void kernel_launch(void* const* d_in, const int* in_sizes, int n_in,
                              void* d_out, int out_size, void* d_ws, size_t ws_size,
                              hipStream_t stream) {
    (void)in_sizes; (void)n_in; (void)out_size; (void)ws_size;
    const float* Xi   = (const float*)d_in[0];
    const float* Si   = (const float*)d_in[1];
    const float* Wq   = (const float*)d_in[2];
    const float* Wk   = (const float*)d_in[3];
    const float* Wv   = (const float*)d_in[4];
    const float* Wout = (const float*)d_in[5];
    const float* W    = (const float*)d_in[6];
    const float* Win  = (const float*)d_in[7];
    const float* bias = (const float*)d_in[8];
    const float* sr   = (const float*)d_in[9];
    const float* lr   = (const float*)d_in[10];

    float* out  = (float*)d_out;
    float* Yi   = out;                       // [B, O]
    float* Snew = out + (size_t)BB * OO;     // [B, M, R]

    // workspace layout (floats)
    float* ws    = (float*)d_ws;
    float* scale = ws;                                   // 8
    float* Qbuf  = ws + 64;                              // B*D      = 131072
    float* Kbuf  = Qbuf + (size_t)BB * DD;               // M*B*D    = 1048576
    float* Vbuf  = Kbuf + (size_t)MM * BB * DD;          // M*B*D    = 1048576
    float* Uibuf = Vbuf + (size_t)MM * BB * DD;          // B*M*D    = 1048576

    esn_spectral_kernel<<<dim3(MM), dim3(256), 0, stream>>>(W, sr, scale);

    esn_q_kernel<<<dim3(BB / 64, DD / 32), dim3(128), 0, stream>>>(Xi, Wq, Qbuf);

    esn_kv_kernel<<<dim3(BB / 64, DD / 16, MM), dim3(128), 0, stream>>>(
        Si, Wk, Wv, Kbuf, Vbuf);

    esn_attn_kernel<<<dim3((BB * MM * MM) / 256), dim3(256), 0, stream>>>(
        Qbuf, Kbuf, Vbuf, Uibuf);

    esn_update_kernel<<<dim3(BB / 64, RR / 32, MM), dim3(128), 0, stream>>>(
        Si, Uibuf, Win, W, scale, bias, lr, Snew);

    esn_out_kernel<<<dim3(BB / 64, OO / 32), dim3(128), 0, stream>>>(
        Snew, Wout, Yi);
}
